// GCN_50981261804182
// MI455X (gfx1250) — compile-verified, matrix operand order checked by reading, not run on hardware
//
#include <hip/hip_runtime.h>

typedef __attribute__((ext_vector_type(16))) __bf16 v16bf;
typedef __attribute__((ext_vector_type(8)))  float  v8f;

#define BM 128
#define BN 64
#define KT 32
#define LDSP (KT + 2)   // padded LDS row stride (even -> 4B-aligned pair reads)
#define BN_EPS 1e-5f

union Frag {
    v16bf v;
    unsigned u[8];
};

__device__ __forceinline__ unsigned pack_bf16(float a, float b) {
    union { __bf16 h[2]; unsigned u; } p;
    p.h[0] = (__bf16)a;
    p.h[1] = (__bf16)b;
    return p.u;
}

// Branchless staged copy: A tile (BM x KT) f32 -> bf16 LDS.
// 32-bit offsets off the uniform base. lda even, K even.
__device__ __forceinline__ void stage_A(const float* __restrict__ A, int M, int K, int lda,
                                        int rowBase, int k0, __bf16 (*As)[LDSP], int tid)
{
    const int cq = (tid & 7) * 4;      // k offset within tile: 0..28
    const int r0 = tid >> 3;           // 0..31
    #pragma unroll
    for (int p = 0; p < 4; ++p) {
        const int r    = r0 + p * 32;
        const int gr   = rowBase + r;
        const int grc  = gr < M ? gr : M - 1;    // clamp row
        const int base = grc * lda;              // 32-bit: fits all shapes here
        const bool rin = gr < M;
        #pragma unroll
        for (int c = 0; c < 2; ++c) {
            int gk  = k0 + cq + c * 2;
            int gkc = gk < K - 1 ? gk : K - 2;   // clamp pair base (K even)
            float2 v = *(const float2*)(A + base + gkc);
            float x0 = (rin && gk     < K) ? v.x : 0.0f;
            float x1 = (rin && gk + 1 < K) ? v.y : 0.0f;
            *(unsigned*)&As[r][cq + c * 2] = pack_bf16(x0, x1);
        }
    }
}

// Branchless staged copy: B tile transposed -> Bs[n][k] = B[(k0+k)*N + colBase+n].
__device__ __forceinline__ void stage_B(const float* __restrict__ B, int K, int N,
                                        int colBase, int k0, __bf16 (*Bs)[LDSP], int tid)
{
    const int n   = tid & 63;
    const int kq  = (tid >> 6) * 8;    // 0,8,16,24
    const int gn  = colBase + n;
    const int gnc = gn < N ? gn : N - 1;
    const bool nin = gn < N;
    #pragma unroll
    for (int i = 0; i < 8; i += 2) {
        int gk0 = k0 + kq + i;
        int gk1 = gk0 + 1;
        int gc0 = gk0 < K ? gk0 : K - 1;
        int gc1 = gk1 < K ? gk1 : K - 1;
        float a = B[gc0 * N + gnc];
        float b = B[gc1 * N + gnc];
        a = (nin && gk0 < K) ? a : 0.0f;
        b = (nin && gk1 < K) ? b : 0.0f;
        *(unsigned*)&Bs[n][kq + i] = pack_bf16(a, b);
    }
}

// ---------------------------------------------------------------------------
// Generic dual-GEMM + epilogue:
//   C[M,N] = act( BN( A1@B1 + A2@B2 + D + bias ) )
// A row-major f32 [M,lda], B row-major f32 [K,N]. Converted to bf16 in LDS
// (double-buffered), computed with v_wmma_f32_16x16x32_bf16 (f32 accumulate).
// ---------------------------------------------------------------------------
__global__ __launch_bounds__(256)
void gemm_bf16_wmma(const float* __restrict__ A1, const float* __restrict__ B1, int K1, int lda1,
                    const float* __restrict__ A2, const float* __restrict__ B2, int K2, int lda2,
                    const float* __restrict__ Dadd, const float* __restrict__ bias,
                    const float* __restrict__ bnG, const float* __restrict__ bnB,
                    const float* __restrict__ bnM, const float* __restrict__ bnV,
                    int relu, float* __restrict__ Cout, int M, int N)
{
    __shared__ __bf16 As[2][BM][LDSP];
    __shared__ __bf16 Bs[2][BN][LDSP];

    const int tid  = threadIdx.x;
    const int lane = tid & 31;
    const int wv   = tid >> 5;           // 0..7 -> 16-row M sub-tile
    const int hi   = lane >> 4;          // lane half selects K sub-pattern
    const int lm   = lane & 15;
    const int rowBase = blockIdx.y * BM;
    const int colBase = blockIdx.x * BN;

    v8f acc[4];
    #pragma unroll
    for (int nt = 0; nt < 4; ++nt)
        #pragma unroll
        for (int e = 0; e < 8; ++e) acc[nt][e] = 0.0f;

    const int nk1 = (A1 && B1) ? (K1 + KT - 1) / KT : 0;
    const int nk2 = (A2 && B2) ? (K2 + KT - 1) / KT : 0;
    const int nk  = nk1 + nk2;

    if (nk > 0) {
        // prologue: stage chunk 0 into buffer 0
        if (nk1) {
            stage_A(A1, M, K1, lda1, rowBase, 0, As[0], tid);
            stage_B(B1, K1, N, colBase, 0, Bs[0], tid);
        } else {
            stage_A(A2, M, K2, lda2, rowBase, 0, As[0], tid);
            stage_B(B2, K2, N, colBase, 0, Bs[0], tid);
        }
        __syncthreads();

        for (int i = 0; i < nk; ++i) {
            const int cur = i & 1;
            const int nxt = cur ^ 1;

            // issue global loads for the next chunk before computing this one
            const int j = i + 1;
            if (j < nk) {
                if (j < nk1) {          // uniform branch
                    stage_A(A1, M, K1, lda1, rowBase, j * KT, As[nxt], tid);
                    stage_B(B1, K1, N, colBase, j * KT, Bs[nxt], tid);
                } else {
                    stage_A(A2, M, K2, lda2, rowBase, (j - nk1) * KT, As[nxt], tid);
                    stage_B(B2, K2, N, colBase, (j - nk1) * KT, Bs[nxt], tid);
                }
            }

            // ---- build fragments per documented 16-bit layout ----
            // lanes 0-15: V0-3 => K = 0..7,  V4-7 => K = 16..23
            // lanes16-31: V0-3 => K = 8..15, V4-7 => K = 24..31
            Frag a;
            #pragma unroll
            for (int v = 0; v < 8; ++v) {
                int kb = (v < 4) ? (hi * 8 + 2 * v) : (16 + hi * 8 + 2 * (v - 4));
                a.u[v] = *(const unsigned*)&As[cur][wv * 16 + lm][kb];
            }
            #pragma unroll
            for (int nt = 0; nt < 4; ++nt) {
                Frag b;
                #pragma unroll
                for (int v = 0; v < 8; ++v) {
                    int kb = (v < 4) ? (hi * 8 + 2 * v) : (16 + hi * 8 + 2 * (v - 4));
                    b.u[v] = *(const unsigned*)&Bs[cur][nt * 16 + lm][kb];
                }
                acc[nt] = __builtin_amdgcn_wmma_f32_16x16x32_bf16(
                    false, a.v, false, b.v, (short)0, acc[nt], false, false);
            }
            __syncthreads();
        }
    }

    // ---- epilogue: D/bias add, folded BN, ReLU, store f32 (32-bit offsets) ----
    // C/D layout: VGPR r -> M = r + 8*hi ; lane lm -> N
    #pragma unroll
    for (int nt = 0; nt < 4; ++nt) {
        int gc = colBase + nt * 16 + lm;
        if (gc >= N) continue;
        float bia = bias ? bias[gc] : 0.0f;
        float sc = 1.0f, sh = 0.0f;
        if (bnG) {
            float inv = rsqrtf(bnV[gc] + BN_EPS);
            sc = bnG[gc] * inv;
            sh = bnB[gc] - bnM[gc] * sc;
        }
        #pragma unroll
        for (int r = 0; r < 8; ++r) {
            int gr = rowBase + wv * 16 + r + 8 * hi;
            if (gr >= M) continue;
            float val = acc[nt][r] + bia;
            if (Dadd) val += Dadd[gr * N + gc];
            val = val * sc + sh;
            if (relu) val = fmaxf(val, 0.0f);
            Cout[gr * N + gc] = val;
        }
    }
}

// ---------------------------------------------------------------------------
__global__ void zero_f32(float* p, int n) {
    int i = blockIdx.x * blockDim.x + threadIdx.x;
    if (i < n) p[i] = 0.0f;
}

// agg[dst[e]] += feat[src[e]]  (F = 128, float4 per thread chunk)
__global__ void scatter_add_f32(const float* __restrict__ feat,
                                const int* __restrict__ src,
                                const int* __restrict__ dst,
                                float* __restrict__ agg, int E, int F)
{
    int idx = blockIdx.x * blockDim.x + threadIdx.x;
    int per = F >> 2;
    int e = idx / per;
    if (e >= E) return;
    int c = (idx - e * per) << 2;
    int s = src[e], d = dst[e];
    const float4 v = *(const float4*)(feat + (size_t)s * F + c);
    float* q = agg + (size_t)d * F + c;
    atomicAdd(q + 0, v.x);
    atomicAdd(q + 1, v.y);
    atomicAdd(q + 2, v.z);
    atomicAdd(q + 3, v.w);
}

// ---------------------------------------------------------------------------
extern "C" void kernel_launch(void* const* d_in, const int* in_sizes, int n_in,
                              void* d_out, int out_size, void* d_ws, size_t ws_size,
                              hipStream_t stream)
{
    const int N0 = 200000, N1 = 40000, N2 = 8192, N3 = 2048;
    const int E0 = 400000, E1 = 81920, E2 = 20480;
    const int F_IN = 602, H = 128, H2 = 64, C = 41;

    const float* x     = (const float*)d_in[0];
    const int*   src0  = (const int*)d_in[1];
    const int*   dst0  = (const int*)d_in[2];
    const int*   src1  = (const int*)d_in[3];
    const int*   dst1  = (const int*)d_in[4];
    const int*   src2  = (const int*)d_in[5];
    const int*   dst2  = (const int*)d_in[6];

    const float* w_rel0 = (const float*)d_in[7];
    const float* b_rel0 = (const float*)d_in[8];
    const float* w_root0= (const float*)d_in[9];
    const float* g0 = (const float*)d_in[10], *bb0 = (const float*)d_in[11];
    const float* m0 = (const float*)d_in[12], *v0  = (const float*)d_in[13];

    const float* w_rel1 = (const float*)d_in[14];
    const float* b_rel1 = (const float*)d_in[15];
    const float* w_root1= (const float*)d_in[16];
    const float* g1 = (const float*)d_in[17], *bb1 = (const float*)d_in[18];
    const float* m1p= (const float*)d_in[19], *v1  = (const float*)d_in[20];

    const float* w_rel2 = (const float*)d_in[21];
    const float* b_rel2 = (const float*)d_in[22];
    const float* w_root2= (const float*)d_in[23];
    const float* g2 = (const float*)d_in[24], *bb2 = (const float*)d_in[25];
    const float* m2 = (const float*)d_in[26], *v2  = (const float*)d_in[27];

    const float* w1f = (const float*)d_in[28];
    const float* b1f = (const float*)d_in[29];
    const float* w2f = (const float*)d_in[30];
    const float* b2f = (const float*)d_in[31];

    float* out = (float*)d_out;

    // workspace layout (f32 elements), with reuse; ~123 MB total
    float* ws   = (float*)d_ws;
    float* y0   = ws;                          // [N0,128] pre-transformed feats
    float* h1   = ws;                          // reuses y0 region (y0 dead)
    float* agg0 = ws + (size_t)N0 * H;         // [N1,128]
    float* agg1 = agg0;                        // reuses agg0 (dead)
    float* h2   = agg0 + (size_t)N2 * H;       // [N2,128]
    float* agg2 = agg0 + (size_t)2 * N2 * H;   // [N3,128]
    float* h3   = agg2 + (size_t)N3 * H;       // [N3,128]
    float* mlp1 = h3   + (size_t)N3 * H;       // [N3,64]

    dim3 blk(256);
    auto cdiv = [](int a, int b) { return (a + b - 1) / b; };

    // 1) y0 = x @ w_rel0   (segment_sum commutes with linear map)
    gemm_bf16_wmma<<<dim3(cdiv(H, BN), cdiv(N0, BM)), blk, 0, stream>>>(
        x, w_rel0, F_IN, F_IN, nullptr, nullptr, 0, 0,
        nullptr, nullptr, nullptr, nullptr, nullptr, nullptr,
        0, y0, N0, H);

    // 2) agg0 = segment_sum(y0[src0] -> dst0)
    zero_f32<<<cdiv(N1 * H, 256), blk, 0, stream>>>(agg0, N1 * H);
    scatter_add_f32<<<cdiv(E0 * (H / 4), 256), blk, 0, stream>>>(y0, src0, dst0, agg0, E0, H);

    // 3) h1 = ReLU(BN(x[:N1] @ w_root0 + agg0 + b_rel0))
    gemm_bf16_wmma<<<dim3(cdiv(H, BN), cdiv(N1, BM)), blk, 0, stream>>>(
        x, w_root0, F_IN, F_IN, nullptr, nullptr, 0, 0,
        agg0, b_rel0, g0, bb0, m0, v0, 1, h1, N1, H);

    // 4) agg1 = segment_sum(h1[src1] -> dst1)
    zero_f32<<<cdiv(N2 * H, 256), blk, 0, stream>>>(agg1, N2 * H);
    scatter_add_f32<<<cdiv(E1 * (H / 4), 256), blk, 0, stream>>>(h1, src1, dst1, agg1, E1, H);

    // 5) h2 = ReLU(BN(agg1 @ w_rel1 + h1[:N2] @ w_root1 + b_rel1))
    gemm_bf16_wmma<<<dim3(cdiv(H, BN), cdiv(N2, BM)), blk, 0, stream>>>(
        agg1, w_rel1, H, H, h1, w_root1, H, H,
        nullptr, b_rel1, g1, bb1, m1p, v1, 1, h2, N2, H);

    // 6) agg2 = segment_sum(h2[src2] -> dst2)
    zero_f32<<<cdiv(N3 * H, 256), blk, 0, stream>>>(agg2, N3 * H);
    scatter_add_f32<<<cdiv(E2 * (H / 4), 256), blk, 0, stream>>>(h2, src2, dst2, agg2, E2, H);

    // 7) h3 = ReLU(BN(agg2 @ w_rel2 + h2[:N3] @ w_root2 + b_rel2))
    gemm_bf16_wmma<<<dim3(cdiv(H, BN), cdiv(N3, BM)), blk, 0, stream>>>(
        agg2, w_rel2, H, H, h2, w_root2, H, H,
        nullptr, b_rel2, g2, bb2, m2, v2, 1, h3, N3, H);

    // 8) mlp1 = ReLU(h3 @ w1 + b1)
    gemm_bf16_wmma<<<dim3(cdiv(H2, BN), cdiv(N3, BM)), blk, 0, stream>>>(
        h3, w1f, H, H, nullptr, nullptr, 0, 0,
        nullptr, b1f, nullptr, nullptr, nullptr, nullptr, 1, mlp1, N3, H2);

    // 9) out = mlp1 @ w2 + b2
    gemm_bf16_wmma<<<dim3(cdiv(C, BN), cdiv(N3, BM)), blk, 0, stream>>>(
        mlp1, w2f, H2, H2, nullptr, nullptr, 0, 0,
        nullptr, b2f, nullptr, nullptr, nullptr, nullptr, 0, out, N3, C);

    (void)in_sizes; (void)n_in; (void)out_size; (void)ws_size;
}
